// GATEncoder_33947421507637
// MI455X (gfx1250) — matrix operand order repaired and checked by reading, not compiled, not run on hardware
//
#include <hip/hip_runtime.h>

typedef __attribute__((ext_vector_type(16))) _Float16 v16h;
typedef __attribute__((ext_vector_type(8)))  _Float16 v8h;
typedef __attribute__((ext_vector_type(8)))  float    v8f;

static constexpr int NN   = 50000;
static constexpr int EE   = 800000;
static constexpr int FD   = 128;   // feature dim everywhere (IN = HEADS*HID = OUT = 128)
static constexpr int GG   = 64;
static constexpr float NEG_SLOPE = 0.2f;

// ---------- float ordered-uint encoding for atomic max ----------
__device__ __forceinline__ unsigned fenc(float f) {
    unsigned u = __float_as_uint(f);
    return (u & 0x80000000u) ? ~u : (u | 0x80000000u);
}
__device__ __forceinline__ float fdec(unsigned u) {
    unsigned b = (u & 0x80000000u) ? (u & 0x7FFFFFFFu) : ~u;
    return __uint_as_float(b);
}

// =====================================================================
// Convert f32 -> f16, 8 elements per thread (for the layer-1 input x)
// =====================================================================
__global__ void cvt_f32_to_f16(const float* __restrict__ in, _Float16* __restrict__ out,
                               int nelem) {
    int t = blockIdx.x * blockDim.x + threadIdx.x;
    int base = t * 8;
    if (base >= nelem) return;
    float4 f0 = *(const float4*)(in + base);
    float4 f1 = *(const float4*)(in + base + 4);
    v8h o;
    o[0] = (_Float16)f0.x; o[1] = (_Float16)f0.y; o[2] = (_Float16)f0.z; o[3] = (_Float16)f0.w;
    o[4] = (_Float16)f1.x; o[5] = (_Float16)f1.y; o[6] = (_Float16)f1.z; o[7] = (_Float16)f1.w;
    *(v8h*)(out + base) = o;
}

// =====================================================================
// Pack W[128,128] f32 (row-major, K x Ncol) into per-lane WMMA B-fragment
// layout, f16:  P[((ct*4 + ks)*32 + lane)*16 + j] = W[ks*32 + offB + j, ct*16 + lane%16]
// where offB = (lane<16) ? 0 : 16.   1024 threads total.
// =====================================================================
__global__ void pack_w_f16(const float* __restrict__ W, _Float16* __restrict__ P) {
    int t = blockIdx.x * blockDim.x + threadIdx.x;
    if (t >= 1024) return;
    int lane = t & 31;
    int ks   = (t >> 5) & 3;
    int ct   = t >> 7;
    int col  = ct * 16 + (lane & 15);
    int offB = (lane < 16) ? 0 : 16;
    _Float16* dst = P + (size_t)t * 16;
#pragma unroll
    for (int j = 0; j < 16; ++j)
        dst[j] = (_Float16)W[(size_t)(ks * 32 + offB + j) * FD + col];
}

// =====================================================================
// WMMA GEMM: C[N,128] = A[N,128](f16) @ W[128,128] (+ bias optional)
// blockDim=256 (8 waves). Block stages the 16x128 f16 A-strip (4KB,
// contiguous) into LDS with GLOBAL_LOAD_ASYNC_TO_LDS_B128 (one 16B
// chunk per thread), waits on ASYNCcnt, barriers, then wave w computes
// column tile w with packed-B 32B loads. grid.x = N/16 exactly (3125).
// =====================================================================
__global__ void gemm128_wmma(const _Float16* __restrict__ Ah,
                             const _Float16* __restrict__ P,   // packed B frags
                             const float* __restrict__ bias,
                             float* __restrict__ C, int nrows) {
    __shared__ _Float16 sA[16 * FD];   // 4 KB

    const int rowbase = blockIdx.x * 16;

    // ---- async stage: strip is contiguous (row stride = 128 halves = 256B)
    {
        const unsigned t = threadIdx.x;                       // 0..255
        unsigned lds_off = (unsigned)(uintptr_t)sA + t * 16;  // addrspace(3) offset
        const char* gsrc = (const char*)Ah + (size_t)rowbase * FD * sizeof(_Float16)
                         + (size_t)t * 16;
        asm volatile("global_load_async_to_lds_b128 %0, %1, off"
                     :: "v"(lds_off), "v"(gsrc) : "memory");
        asm volatile("s_wait_asynccnt 0" ::: "memory");
    }
    __syncthreads();

    const int lane = threadIdx.x & 31;
    const int wv   = threadIdx.x >> 5;     // 0..7 -> column tile
    const int l16  = lane & 15;
    const int hi   = (lane >= 16);
    const int col  = wv * 16 + l16;
    const int offA = hi ? 16 : 0;          // contiguous K order, matches packed B

    v8f c;
    const float bv = bias ? bias[col] : 0.0f;
#pragma unroll
    for (int r = 0; r < 8; ++r) c[r] = bv;

    const v16h* __restrict__ bbase = (const v16h*)P + (size_t)wv * 128 + lane;

#pragma unroll
    for (int kk = 0; kk < 4; ++kk) {
        const int ks = kk * 32;
        // A fragment from LDS: 16 contiguous halves at K=ks+offA
        const _Float16* ap = sA + l16 * FD + ks + offA;
        v8h lo  = *(const v8h*)ap;
        v8h hi8 = *(const v8h*)(ap + 8);
        v16h af = __builtin_shufflevector(lo, hi8, 0, 1, 2, 3, 4, 5, 6, 7,
                                          8, 9, 10, 11, 12, 13, 14, 15);
        // packed B fragment: one 32-byte load
        v16h bf = bbase[kk * 32];

        c = __builtin_amdgcn_wmma_f32_16x16x32_f16(
                /*neg_a=*/false, af, /*neg_b=*/false, bf,
                /*c_mod=*/(short)0, c, /*reuse_a=*/false, /*reuse_b=*/false);
    }

    // ---- store: VGPR r -> row rowbase + r + (hi?8:0), col fixed per lane
    float* __restrict__ crow = C + (size_t)(rowbase + (hi ? 8 : 0)) * FD + col;
#pragma unroll
    for (int r = 0; r < 8; ++r) crow[(size_t)r * FD] = c[r];
}

// =====================================================================
// Attention logits: al_s[n,h] = <h[n,h,:], a_src[h,:]>, same for al_d
// =====================================================================
__global__ void attn_logits(const float* __restrict__ hT,
                            const float* __restrict__ a_src,
                            const float* __restrict__ a_dst,
                            float* __restrict__ als, float* __restrict__ ald,
                            int n, int heads, int chan) {
    int t = blockIdx.x * blockDim.x + threadIdx.x;
    if (t >= n * heads) return;
    int node = t / heads, hd = t - node * heads;
    const float* hp = hT + (size_t)node * FD + hd * chan;
    const float* as = a_src + hd * chan;
    const float* ad = a_dst + hd * chan;
    float s = 0.f, d = 0.f;
    for (int c = 0; c < chan; ++c) { float v = hp[c]; s += v * as[c]; d += v * ad[c]; }
    als[t] = s; ald[t] = d;
}

__device__ __forceinline__ void edge_sd(const int* __restrict__ ei, int e, int E,
                                        int& s, int& d) {
    if (e < E) { s = ei[e]; d = ei[E + e]; } else { s = d = e - E; }   // self loops
}

// =====================================================================
// Pass 1: segment max of leaky-relu(e) per destination, per head
// =====================================================================
__global__ void edge_max(const int* __restrict__ ei, int E, int n, int heads,
                         const float* __restrict__ als, const float* __restrict__ ald,
                         unsigned* __restrict__ menc) {
    int e = blockIdx.x * blockDim.x + threadIdx.x;
    if (e >= E + n) return;
    int s, d; edge_sd(ei, e, E, s, d);
    for (int h = 0; h < heads; ++h) {
        float v = als[s * heads + h] + ald[d * heads + h];
        v = (v >= 0.f) ? v : NEG_SLOPE * v;
        atomicMax(&menc[d * heads + h], fenc(v));
    }
}

// =====================================================================
// Pass 2: segment sum of exp(e - max)
// =====================================================================
__global__ void edge_expsum(const int* __restrict__ ei, int E, int n, int heads,
                            const float* __restrict__ als, const float* __restrict__ ald,
                            const unsigned* __restrict__ menc, float* __restrict__ ssum) {
    int e = blockIdx.x * blockDim.x + threadIdx.x;
    if (e >= E + n) return;
    int s, d; edge_sd(ei, e, E, s, d);
    for (int h = 0; h < heads; ++h) {
        float v = als[s * heads + h] + ald[d * heads + h];
        v = (v >= 0.f) ? v : NEG_SLOPE * v;
        float p = __expf(v - fdec(menc[d * heads + h]));
        atomicAdd(&ssum[d * heads + h], p);
    }
}

// =====================================================================
// Pass 3: agg[dst, :] += alpha * hT[src, :]   (32 lanes/edge, 4 ch each)
// =====================================================================
__global__ void edge_aggregate(const int* __restrict__ ei, int E, int n, int heads, int chan,
                               const float* __restrict__ als, const float* __restrict__ ald,
                               const unsigned* __restrict__ menc, const float* __restrict__ ssum,
                               const float* __restrict__ hT, float* __restrict__ agg) {
    unsigned t = blockIdx.x * blockDim.x + threadIdx.x;
    unsigned tot = (unsigned)(E + n) * 32u;
    if (t >= tot) return;
    int lane = t & 31;
    int e = (int)(t >> 5);
    int s, d; edge_sd(ei, e, E, s, d);
    int c0 = lane * 4;
    int hd = c0 / chan;
    float v = als[s * heads + hd] + ald[d * heads + hd];
    v = (v >= 0.f) ? v : NEG_SLOPE * v;
    float alpha = __expf(v - fdec(menc[d * heads + hd])) / ssum[d * heads + hd];
    const float4 hv = *(const float4*)(hT + (size_t)s * FD + c0);
    float* ap = agg + (size_t)d * FD + c0;
    atomicAdd(ap + 0, hv.x * alpha);
    atomicAdd(ap + 1, hv.y * alpha);
    atomicAdd(ap + 2, hv.z * alpha);
    atomicAdd(ap + 3, hv.w * alpha);
}

// =====================================================================
// Epilogues.  act is stored directly as f16 (only the next GEMM reads it).
// =====================================================================
__global__ void ep_relu_skip(const float* __restrict__ agg, const float* __restrict__ b,
                             const float* __restrict__ xskip, _Float16* __restrict__ act,
                             int n) {
    int t = blockIdx.x * blockDim.x + threadIdx.x;
    if (t >= n * FD) return;
    int c = t & (FD - 1);
    float v = agg[t] + b[c] + xskip[t];
    act[t] = (_Float16)(v > 0.f ? v : 0.f);
}

__global__ void ep3_pool(const float* __restrict__ agg, const float* __restrict__ b3,
                         const int* __restrict__ batch, float* __restrict__ hout,
                         float* __restrict__ ctx, int n) {
    int t = blockIdx.x * blockDim.x + threadIdx.x;
    if (t >= n * FD) return;
    int node = t >> 7, c = t & (FD - 1);
    float v = agg[t] + b3[c];
    hout[t] = v;
    atomicAdd(&ctx[(size_t)batch[node] * FD + c], v);
}

__global__ void count_nodes(const int* __restrict__ batch, float* __restrict__ cnt, int n) {
    int t = blockIdx.x * blockDim.x + threadIdx.x;
    if (t < n) atomicAdd(&cnt[batch[t]], 1.0f);
}

__global__ void pool_div(float* __restrict__ ctx, const float* __restrict__ cnt) {
    int t = blockIdx.x * blockDim.x + threadIdx.x;
    if (t >= GG * FD) return;
    ctx[t] /= fmaxf(cnt[t >> 7], 1.0f);
}

// =====================================================================
extern "C" void kernel_launch(void* const* d_in, const int* in_sizes, int n_in,
                              void* d_out, int out_size, void* d_ws, size_t ws_size,
                              hipStream_t stream) {
    const float* x     = (const float*)d_in[0];
    const int*   ei    = (const int*)d_in[1];
    const int*   batch = (const int*)d_in[2];
    const float* Wp = (const float*)d_in[3];  const float* bp = (const float*)d_in[4];
    const float* W1 = (const float*)d_in[5];  const float* as1 = (const float*)d_in[6];
    const float* ad1 = (const float*)d_in[7]; const float* b1 = (const float*)d_in[8];
    const float* W2 = (const float*)d_in[9];  const float* as2 = (const float*)d_in[10];
    const float* ad2 = (const float*)d_in[11];const float* b2 = (const float*)d_in[12];
    const float* W3 = (const float*)d_in[13]; const float* as3 = (const float*)d_in[14];
    const float* ad3 = (const float*)d_in[15];const float* b3 = (const float*)d_in[16];

    auto up = [](size_t v) { return (v + 255) & ~(size_t)255; };
    const size_t NF  = (size_t)NN * FD * sizeof(float);
    const size_t NH  = (size_t)NN * FD * sizeof(_Float16);
    const size_t PW  = (size_t)FD * FD * sizeof(_Float16);   // 32KB packed weights
    char* p = (char*)d_ws;
    float*    xskip = (float*)p;     p += up(NF);
    float*    hT    = (float*)p;     p += up(NF);
    float*    agg   = (float*)p;     p += up(NF);
    _Float16* xh    = (_Float16*)p;  p += up(NH);
    _Float16* act   = (_Float16*)p;  p += up(NH);
    _Float16* Pp    = (_Float16*)p;  p += up(PW);
    _Float16* P1    = (_Float16*)p;  p += up(PW);
    _Float16* P2    = (_Float16*)p;  p += up(PW);
    _Float16* P3    = (_Float16*)p;  p += up(PW);
    float*    als   = (float*)p;     p += up((size_t)NN * 4 * sizeof(float));
    float*    ald   = (float*)p;     p += up((size_t)NN * 4 * sizeof(float));
    unsigned* menc  = (unsigned*)p;  p += up((size_t)NN * 4 * sizeof(unsigned));
    float*    ssum  = (float*)p;     p += up((size_t)NN * 4 * sizeof(float));
    float*    cnt   = (float*)p;     p += up((size_t)GG * sizeof(float));

    float* hout = (float*)d_out;
    float* ctx  = hout + (size_t)NN * FD;

    const int ETOT = EE + NN;
    const dim3 B256(256);
    const int gGemm = NN / 16;                          // 3125, exact
    const int gEdge = (ETOT + 255) / 256;
    const int gAgg  = (int)(((size_t)ETOT * 32 + 255) / 256);
    const int gFeat = (NN * FD + 255) / 256;
    const int gCvt  = (NN * FD / 8 + 255) / 256;

    // ---------- one-time preprocessing ----------
    cvt_f32_to_f16<<<gCvt, B256, 0, stream>>>(x, xh, NN * FD);
    pack_w_f16<<<4, B256, 0, stream>>>(Wp, Pp);
    pack_w_f16<<<4, B256, 0, stream>>>(W1, P1);
    pack_w_f16<<<4, B256, 0, stream>>>(W2, P2);
    pack_w_f16<<<4, B256, 0, stream>>>(W3, P3);

    // ---------- input skip projection ----------
    gemm128_wmma<<<gGemm, B256, 0, stream>>>(xh, Pp, bp, xskip, NN);

    const _Float16* layer_in[3] = { xh, act, act };
    const _Float16* Ps[3]  = { P1, P2, P3 };
    const float*    asv[3] = { as1, as2, as3 };
    const float*    adv[3] = { ad1, ad2, ad3 };
    const int       hdc[3] = { 4, 4, 1 };
    const int       chc[3] = { 32, 32, 128 };

    for (int L = 0; L < 3; ++L) {
        const int H = hdc[L], C = chc[L];
        gemm128_wmma<<<gGemm, B256, 0, stream>>>(layer_in[L], Ps[L], nullptr, hT, NN);
        attn_logits<<<(NN * H + 255) / 256, B256, 0, stream>>>(hT, asv[L], adv[L], als, ald, NN, H, C);
        hipMemsetAsync(menc, 0, (size_t)NN * H * sizeof(unsigned), stream);
        hipMemsetAsync(ssum, 0, (size_t)NN * H * sizeof(float), stream);
        hipMemsetAsync(agg,  0, NF, stream);
        edge_max   <<<gEdge, B256, 0, stream>>>(ei, EE, NN, H, als, ald, menc);
        edge_expsum<<<gEdge, B256, 0, stream>>>(ei, EE, NN, H, als, ald, menc, ssum);
        edge_aggregate<<<gAgg, B256, 0, stream>>>(ei, EE, NN, H, C, als, ald, menc, ssum, hT, agg);
        if (L == 0) {
            ep_relu_skip<<<gFeat, B256, 0, stream>>>(agg, b1, xskip, act, NN);
        } else if (L == 1) {
            ep_relu_skip<<<gFeat, B256, 0, stream>>>(agg, b2, xskip, act, NN);
        } else {
            hipMemsetAsync(ctx, 0, (size_t)GG * FD * sizeof(float), stream);
            hipMemsetAsync(cnt, 0, (size_t)GG * sizeof(float), stream);
            ep3_pool<<<gFeat, B256, 0, stream>>>(agg, b3, batch, hout, ctx, NN);
            count_nodes<<<(NN + 255) / 256, B256, 0, stream>>>(batch, cnt, NN);
            pool_div<<<(GG * FD + 255) / 256, B256, 0, stream>>>(ctx, cnt);
        }
    }
}
